// PINN_29051158790862
// MI455X (gfx1250) — matrix operand order repaired and checked by reading, not compiled, not run on hardware
//
#include <hip/hip_runtime.h>
#include <cstdint>

typedef __attribute__((ext_vector_type(16))) _Float16 v16h;
typedef __attribute__((ext_vector_type(8)))  _Float16 v8h;
typedef __attribute__((ext_vector_type(8)))  float    v8f;

#define HID 256
#define SPB 16              // samples per block
#define ROWS (SPB * 7)      // 112 jet rows: {val, dx, dy, dt, dxx, dxy, dyy}
#define HS 264              // f16 elements per LDS row (16B-aligned K-chunks)

__device__ __forceinline__ v16h cat8(v8h lo, v8h hi) {
  v16h r;
#pragma unroll
  for (int i = 0; i < 8; ++i) { r[i] = lo[i]; r[8 + i] = hi[i]; }
  return r;
}

// tanh jet chain rule: h=tanh(z); h_a = s z_a; h_ab = s z_ab - 2 h s z_a z_b
__device__ __forceinline__ void jet_tanh(const float z[7], float h[7]) {
  float hv = tanhf(z[0]);
  float s  = 1.0f - hv * hv;
  float m2 = -2.0f * hv * s;
  h[0] = hv;
  h[1] = s * z[1];
  h[2] = s * z[2];
  h[3] = s * z[3];
  h[4] = s * z[4] + m2 * z[1] * z[1];
  h[5] = s * z[5] + m2 * z[1] * z[2];
  h[6] = s * z[6] + m2 * z[2] * z[2];
}

// ---- transpose W1..W3 (256x256 f32, K-major) -> Wt (N-major f16) ----------
__global__ void pinn_prep(const float* __restrict__ W1, const float* __restrict__ W2,
                          const float* __restrict__ W3, _Float16* __restrict__ Wt) {
  int idx = blockIdx.x * blockDim.x + threadIdx.x;   // 0 .. 3*65536-1
  int l = idx >> 16;
  int r = idx & 65535;
  int n = r >> 8;
  int k = r & 255;
  const float* W = (l == 0) ? W1 : (l == 1) ? W2 : W3;
  Wt[(size_t)l * 65536 + n * 256 + k] = (_Float16)W[k * 256 + n];
}

// ---- main: jet propagation + physics --------------------------------------
__global__ __launch_bounds__(128) void pinn_main(
    const float* __restrict__ W0, const float* __restrict__ b0,
    const float* __restrict__ b1, const float* __restrict__ b2,
    const float* __restrict__ b3,
    const float* __restrict__ W4, const float* __restrict__ b4,
    const float* __restrict__ x, const float* __restrict__ t,
    const _Float16* __restrict__ Wt, float* __restrict__ partials) {
  __shared__ _Float16 Hsm[ROWS * HS];   // activations (GEMM A operand)
  __shared__ _Float16 Zsm[ROWS * HS];   // pre-activations (GEMM D)
  __shared__ float    sol[SPB][3][7];   // output-layer jets
  __shared__ float    red[SPB][5];      // per-sample residual terms

  const int tid  = threadIdx.x;
  const int lane = tid & 31;
  const int hi   = lane >> 4;           // 0: lanes 0-15, 1: lanes 16-31
  const int lm   = lane & 15;
  const int wave = tid >> 5;

  // ---- layer 0 (3 -> 256) + tanh jet, scalar ------------------------------
  for (int idx = tid; idx < SPB * HID; idx += blockDim.x) {
    int s = idx >> 8, j = idx & 255;
    int gs = blockIdx.x * SPB + s;
    float x0 = x[gs * 2 + 0], y = x[gs * 2 + 1], tt = t[gs];
    float w0 = W0[j], w1 = W0[256 + j], w2 = W0[512 + j];
    float z[7] = { x0 * w0 + y * w1 + tt * w2 + b0[j], w0, w1, w2, 0.f, 0.f, 0.f };
    float h[7];
    jet_tanh(z, h);
#pragma unroll
    for (int c = 0; c < 7; ++c) Hsm[(s * 7 + c) * HS + j] = (_Float16)h[c];
  }

  const float* biases[3] = { b1, b2, b3 };
  for (int layer = 0; layer < 3; ++layer) {
    __syncthreads();
    const _Float16* Wl = Wt + (size_t)layer * HID * HID;
    // ---- Z[112x256] = H[112x256] @ W[256x256] via v_wmma_f32_16x16x32_f16.
    // Each wave owns whole N-column panels: the 256x16 f16 B panel is held in
    // 64 VGPRs and reused across all 7 M-tiles (7x less global/L2 traffic).
    for (int nt = wave; nt < HID / 16; nt += 4) {
      const int bn = nt * 16 + lm;
      __builtin_prefetch(Wl + bn * HID, 0, 0);
      v16h B[8];
#pragma unroll
      for (int kb = 0; kb < HID / 32; ++kb) {
        // B (32x16, N-major weights): lanes<16 K=0..15, lanes>=16 K=16..31
        const _Float16* wb = Wl + bn * HID + kb * 32 + hi * 16;
        v8h blo = *reinterpret_cast<const v8h*>(wb);
        v8h bhi = *reinterpret_cast<const v8h*>(wb + 8);
        B[kb] = cat8(blo, bhi);
      }
#pragma unroll
      for (int mt = 0; mt < ROWS / 16; ++mt) {
        const int arow = mt * 16 + lm;
        v8f acc = {};
#pragma unroll
        for (int kb = 0; kb < HID / 32; ++kb) {
          // A (16-bit 16x32): lanes<16 hold K 0..15, lanes>=16 K 16..31
          const _Float16* ha = &Hsm[arow * HS + kb * 32 + hi * 16];
          v8h alo = *reinterpret_cast<const v8h*>(ha);
          v8h ahi = *reinterpret_cast<const v8h*>(ha + 8);
          acc = __builtin_amdgcn_wmma_f32_16x16x32_f16(
              false, cat8(alo, ahi), false, B[kb],
              (short)0, acc, false, false);
        }
        // D layout: VGPR r -> row mt*16 + hi*8 + r, col nt*16 + lm
        int rb = mt * 16 + hi * 8;
#pragma unroll
        for (int r = 0; r < 8; ++r)
          Zsm[(rb + r) * HS + nt * 16 + lm] = (_Float16)acc[r];
      }
    }
    __syncthreads();
    // ---- bias (value channel only) + tanh jet -----------------------------
    const float* bl = biases[layer];
    for (int idx = tid; idx < SPB * HID; idx += blockDim.x) {
      int s = idx >> 8, j = idx & 255;
      float z[7];
#pragma unroll
      for (int c = 0; c < 7; ++c) z[c] = (float)Zsm[(s * 7 + c) * HS + j];
      z[0] += bl[j];
      float h[7];
      jet_tanh(z, h);
#pragma unroll
      for (int c = 0; c < 7; ++c) Hsm[(s * 7 + c) * HS + j] = (_Float16)h[c];
    }
  }
  __syncthreads();

  // ---- output layer (256 -> 3), scalar dot per jet row --------------------
  if (tid < ROWS) {
    int s = tid / 7, c = tid % 7;
    float a0 = (c == 0) ? b4[0] : 0.f;
    float a1 = (c == 0) ? b4[1] : 0.f;
    float a2 = (c == 0) ? b4[2] : 0.f;
    const _Float16* hrow = &Hsm[(s * 7 + c) * HS];
    for (int k = 0; k < HID; ++k) {
      float hv = (float)hrow[k];
      a0 += hv * W4[k * 3 + 0];
      a1 += hv * W4[k * 3 + 1];
      a2 += hv * W4[k * 3 + 2];
    }
    sol[s][0][c] = a0; sol[s][1][c] = a1; sol[s][2][c] = a2;
  }
  __syncthreads();

  // ---- physics per sample -------------------------------------------------
  if (tid < SPB) {
    int gs = blockIdx.x * SPB + tid;
    float y = x[gs * 2 + 1];
    float ld = t[gs];                       // loading(t) = t
    const float MU = 1.0f, KC = 2.0f;       // KC = lambda + 2 mu / dim
    const float GCc = 0.0027f, Lc = 0.02f;

    float S0v = sol[tid][0][0], S0a = sol[tid][0][1], S0b = sol[tid][0][2];
    float S0t = sol[tid][0][3], S0aa = sol[tid][0][4], S0ab = sol[tid][0][5];
    float S0bb = sol[tid][0][6];
    float S1v = sol[tid][1][0], S1a = sol[tid][1][1], S1b = sol[tid][1][2];
    float S1aa = sol[tid][1][4], S1ab = sol[tid][1][5], S1bb = sol[tid][1][6];
    float S2v = sol[tid][2][0], S2a = sol[tid][2][1], S2b = sol[tid][2][2];
    float S2aa = sol[tid][2][4], S2ab = sol[tid][2][5], S2bb = sol[tid][2][6];
    (void)S0ab;

    // phi = sigmoid(S0) jets
    float p   = 1.f / (1.f + expf(-S0v));
    float dp  = p * (1.f - p);
    float ddp = dp * (1.f - 2.f * p);
    float phi_a = dp * S0a, phi_b = dp * S0b, phi_t = dp * S0t;
    float phi_aa = ddp * S0a * S0a + dp * S0aa;
    float phi_bb = ddp * S0b * S0b + dp * S0bb;

    // displacement jets: ux = q(y) S1 ld/3; uy = q(y) S2 ld + y ld; q=y(y-1)
    float q = y * (y - 1.f), qp = 2.f * y - 1.f, qpp = 2.f;
    float c3 = ld * (1.f / 3.f);
    float ux_a  = q * S1a * c3;
    float ux_b  = (qp * S1v + q * S1b) * c3;
    float ux_aa = q * S1aa * c3;
    float ux_ab = (qp * S1a + q * S1ab) * c3;
    float ux_bb = (qpp * S1v + 2.f * qp * S1b + q * S1bb) * c3;
    float uy_a  = q * S2a * ld;
    float uy_b  = (qp * S2v + q * S2b) * ld + ld;
    float uy_aa = q * S2aa * ld;
    float uy_ab = (qp * S2a + q * S2ab) * ld;
    float uy_bb = (qpp * S2v + 2.f * qp * S2b + q * S2bb) * ld;

    // strain and its x-gradients
    float e00 = ux_a, e11 = uy_b, e01 = 0.5f * (ux_b + uy_a);
    float e00a = ux_aa, e11a = uy_ab, e01a = 0.5f * (ux_ab + uy_aa);
    float e00b = ux_ab, e11b = uy_bb, e01b = 0.5f * (ux_bb + uy_ab);

    float tr = e00 + e11, tra = e00a + e11a, trb = e00b + e11b;
    float dd = e00 - e11, dda = e00a - e11a, ddb = e00b - e11b;
    float rp = fmaxf(tr, 0.f),  ip  = (tr > 0.f) ? 1.f : 0.f;
    float rn = fmaxf(-tr, 0.f), in_ = (tr < 0.f) ? 1.f : 0.f;

    float psip = 0.5f * KC * rp * rp + MU * (0.5f * dd * dd + 2.f * e01 * e01);
    float psin = 0.5f * KC * rn * rn;

    // grad psi_pos / psi_neg w.r.t. eps, and their x-derivatives
    float gp00 = KC * rp + MU * dd, gp11 = KC * rp - MU * dd, gp01 = 2.f * MU * e01;
    float gp00a = KC * ip * tra + MU * dda, gp11a = KC * ip * tra - MU * dda;
    float gp01a = 2.f * MU * e01a;
    float gp00b = KC * ip * trb + MU * ddb, gp11b = KC * ip * trb - MU * ddb;
    float gp01b = 2.f * MU * e01b;
    float gn = -KC * rn, gna = KC * in_ * tra, gnb = KC * in_ * trb;

    float omp = 1.f - p;
    float g = omp * omp + 1e-6f, f = omp * omp;
    float ga = -2.f * omp * phi_a, gb = -2.f * omp * phi_b; // d(1-p)^2/dx_k

    float s00 = g * gp00 + gn, s01 = g * gp01, s11 = g * gp11 + gn;
    float s00a = ga * gp00 + g * gp00a + gna;
    float s01a = ga * gp01 + g * gp01a;
    float s01b = gb * gp01 + g * gp01b;
    float s11b = gb * gp11 + g * gp11b + gnb;

    // res_i = sum_j d/dx_j [ (1-p)^2 sigma_ij ]
    float res0 = ga * s00 + f * s00a + gb * s01 + f * s01b;
    float res1 = ga * s01 + f * s01a + gb * s11 + f * s11b;

    // phase-field residual with masks
    float pfr  = GCc * (p / Lc - Lc * (phi_aa + phi_bb)) - 2.f * omp * psip;
    float dphi = phi_t;
    bool mpos = (fabsf(dphi) <= 1e-3f) && (fabsf(p - 1.f) > 1e-3f);
    bool mneg = (fabsf(p - 1.f) <= 1e-3f);
    float pf = mpos ? fmaxf(-pfr, 0.f) : (mneg ? fmaxf(pfr, 0.f) : pfr);

    float energy = omp * 2.f * psip + psin
                 + GCc * (p * p / (2.f * Lc)
                        + 0.5f * Lc * (phi_a * phi_a + phi_b * phi_b));

    red[tid][0] = res0 * res0;
    red[tid][1] = res1 * res1;
    red[tid][2] = pf * pf;
    red[tid][3] = energy;
    red[tid][4] = fmaxf(-dphi, 0.f);
  }
  __syncthreads();

  if (tid < 5) {
    float sum = 0.f;
#pragma unroll
    for (int s = 0; s < SPB; ++s) sum += red[s][tid];
    partials[blockIdx.x * 5 + tid] = sum;
  }
}

// ---- fold per-WG partials into the 4 losses -------------------------------
__global__ void pinn_finalize(const float* __restrict__ partials, int nblk, int n,
                              float* __restrict__ out) {
  if (blockIdx.x != 0 || threadIdx.x != 0) return;
  float a0 = 0.f, a1 = 0.f, a2 = 0.f, a3 = 0.f, a4 = 0.f;
  for (int b = 0; b < nblk; ++b) {
    a0 += partials[b * 5 + 0];
    a1 += partials[b * 5 + 1];
    a2 += partials[b * 5 + 2];
    a3 += partials[b * 5 + 3];
    a4 += partials[b * 5 + 4];
  }
  float invn = 1.0f / (float)n;
  float m0 = a0 * invn, m1 = a1 * invn;
  float mm = 0.5f * (m0 + m1);
  float w0 = mm / (m0 + 1e-6f), w1 = mm / (m1 + 1e-6f);
  out[0] = w0 * m0 + w1 * m1;   // loss_stress (re-weighted)
  out[1] = a2 * invn;           // loss_pf
  out[2] = logf(a3);            // loss_energy
  out[3] = a4 * invn;           // loss_irr
}

extern "C" void kernel_launch(void* const* d_in, const int* in_sizes, int n_in,
                              void* d_out, int out_size, void* d_ws, size_t ws_size,
                              hipStream_t stream) {
  (void)n_in; (void)out_size; (void)ws_size;
  const float* W0 = (const float*)d_in[0];
  const float* b0 = (const float*)d_in[1];
  const float* W1 = (const float*)d_in[2];
  const float* b1 = (const float*)d_in[3];
  const float* W2 = (const float*)d_in[4];
  const float* b2 = (const float*)d_in[5];
  const float* W3 = (const float*)d_in[6];
  const float* b3 = (const float*)d_in[7];
  const float* W4 = (const float*)d_in[8];
  const float* b4 = (const float*)d_in[9];
  const float* x  = (const float*)d_in[10];
  const float* t  = (const float*)d_in[11];

  int N    = in_sizes[10] / 2;          // x is (N, 2)
  int nblk = N / SPB;                   // 512 for N=8192

  _Float16* Wt     = (_Float16*)d_ws;                      // 3 * 256*256 f16
  float* partials  = (float*)((char*)d_ws + (size_t)3 * HID * HID * sizeof(_Float16));

  pinn_prep<<<768, 256, 0, stream>>>(W1, W2, W3, Wt);
  pinn_main<<<nblk, 128, 0, stream>>>(W0, b0, b1, b2, b3, W4, b4, x, t, Wt, partials);
  pinn_finalize<<<1, 1, 0, stream>>>(partials, nblk, N, (float*)d_out);
}
